// TokenQKAttention_88983132438981
// MI455X (gfx1250) — matrix operand order repaired and checked by reading, not compiled, not run on hardware
//
#include <hip/hip_runtime.h>

typedef __attribute__((ext_vector_type(16))) _Float16 v16h;
typedef __attribute__((ext_vector_type(8)))  _Float16 v8h;
typedef __attribute__((ext_vector_type(4)))  _Float16 v4h;
typedef __attribute__((ext_vector_type(8)))  float    v8f;
typedef __attribute__((ext_vector_type(2)))  int      v2i;

#define TT     4
#define BB     64
#define CC     384
#define NN     196
#define NP     224        // padded N (7 * 32)
#define HEADS  8
#define DH     48
#define KTILE  32
#define MTILE  128
#define NTILE  32
#define NBLK   7          // NP / 32
#define MBLK   3          // 384/128
#define KBLK   12         // 384/32
#define LXPAD  40         // padded k-stride for B-staging rows
#define WSWZ   ((size_t)CC * CC)   // elements per swizzled weight matrix

// ---- gfx1250 async global->LDS path (probe-guarded; falls back cleanly) ----
#if defined(__has_builtin)
#if __has_builtin(__builtin_amdgcn_global_load_async_to_lds_b64) && \
    __has_builtin(__builtin_amdgcn_s_wait_asynccnt)
#define HAVE_ASYNC_LDS 1
#endif
#endif

typedef __attribute__((address_space(1))) v2i* g_b64_t;   // global b64 ptr
typedef __attribute__((address_space(3))) v2i* l_b64_t;   // LDS b64 ptr

__device__ __forceinline__ float lif_step(float& v, float pre) {
    v += 0.5f * (pre - v);                 // tau = 2
    float s = (v >= 1.0f) ? 1.0f : 0.0f;   // heaviside(v - 1)
    v *= (1.0f - s);                       // hard reset
    return s;
}

__device__ __forceinline__ v16h cat8(v8h a, v8h b) {
    return __builtin_shufflevector(a, b, 0,1,2,3,4,5,6,7,8,9,10,11,12,13,14,15);
}

// ---------------------------------------------------------------------------
// Prep A: W (fp32 CxC) -> f16 in per-lane A-fragment order:
//   out[ ((((mt*12 + kb)*8 + wr)*32 + L)*16 + e ) ]
// ---------------------------------------------------------------------------
__global__ __launch_bounds__(256)
void swizzle_w(const float* __restrict__ W, _Float16* __restrict__ out)
{
    int o = blockIdx.x * 256 + threadIdx.x;      // 147456 total
    int e  = o & 15;
    int t1 = o >> 4;
    int L  = t1 & 31;
    int t2 = t1 >> 5;
    int wr = t2 & 7;
    int t3 = t2 >> 3;
    int kb = t3 % KBLK;
    int mt = t3 / KBLK;
    int lrow = L & 15, lhi = L >> 4;
    int d = mt * MTILE + wr * 16 + lrow;
    int c = kb * KTILE + lhi * 8 + ((e & 8) ? 16 : 0) + (e & 7);
    out[o] = (_Float16)W[d * CC + c];
}

// ---------------------------------------------------------------------------
// Prep B: x [T,B,C,N] fp32 -> xp [T,B,NP,C] f16 (transposed, n padded w/ 0).
// Tiled 32x32 LDS transpose: coalesced reads and coalesced b64 writes.
// ---------------------------------------------------------------------------
__global__ __launch_bounds__(256)
void xpack_kernel(const float* __restrict__ x, _Float16* __restrict__ xp)
{
    __shared__ _Float16 lt[32][36];       // [c][n], padded
    const int tb  = blockIdx.x;           // t*B + b
    const int nb  = (blockIdx.y % NBLK) * NTILE;
    const int cb  = (blockIdx.y / NBLK) * 32;
    const int tid = threadIdx.x;

    {   // read 32c x 32n tile: 4 n per thread, coalesced over n
        int row  = tid >> 3;              // c offset
        int col4 = (tid & 7) * 4;         // n offset
        v4h h;
#pragma unroll
        for (int i = 0; i < 4; ++i) {
            int n = nb + col4 + i;
            float f = (n < NN) ? x[((size_t)tb * CC + cb + row) * NN + n] : 0.f;
            h[i] = (_Float16)f;
        }
        *(v4h*)&lt[row][col4] = h;
    }
    __syncthreads();
    {   // write transposed: 4 c per thread, coalesced over c
        int nrow = tid >> 3;              // n offset
        int c4   = (tid & 7) * 4;         // c offset
        v4h o;
        o[0] = lt[c4 + 0][nrow];
        o[1] = lt[c4 + 1][nrow];
        o[2] = lt[c4 + 2][nrow];
        o[3] = lt[c4 + 3][nrow];
        *(v4h*)&xp[((size_t)tb * NP + nb + nrow) * CC + cb + c4] = o;
    }
}

// ---------------------------------------------------------------------------
// Kernel 1: q/k GEMMs (f16 WMMA, f32 accum) + fused BN + LIF over T.
// Block 256 thr (8 waves), macro tile M=128 x N=32, K-tile 32, T in 2 passes.
// A-fragments: direct from pre-swizzled global weights (2x b128/lane).
// B-fragments: LDS [tt][n][k], staged via async global->LDS DMA when the
// toolchain exposes it (ASYNCcnt), else a plain b64 copy.
// Spikes stored transposed [t][b][n][c] as v8h (one b128 per subtile).
// ---------------------------------------------------------------------------
__global__ __launch_bounds__(256)
void qk_gemm_lif(const _Float16* __restrict__ xp,
                 const _Float16* __restrict__ wq_sw,
                 const _Float16* __restrict__ wk_sw,
                 const float* __restrict__ qg, const float* __restrict__ qb,
                 const float* __restrict__ qm, const float* __restrict__ qv,
                 const float* __restrict__ kg, const float* __restrict__ kb_,
                 const float* __restrict__ km, const float* __restrict__ kv,
                 _Float16* __restrict__ qs, _Float16* __restrict__ ksp)
{
    __shared__ _Float16 lx[2][NTILE][LXPAD];

    const int tid  = threadIdx.x;
    const int lane = tid & 31;
    const int wv   = tid >> 5;
    const int b    = blockIdx.y;
    const int mt   = blockIdx.x / NBLK;
    const int nt   = blockIdx.x % NBLK;
    const int mbase = mt * MTILE;
    const int nbase = nt * NTILE;
    const int lrow = lane & 15;
    const int lhi  = lane >> 4;
    const int nloc = tid & 31;            // staging: n within tile
    const int k0s  = (tid >> 5) * 4;      // staging: k start (0..28)
    const int n    = nbase + nloc;        // always < NP (in-bounds of xp)

    // folded BN: pre = acc*scale + shift  (row = wv*16 + j + 8*lhi)
    float qscale[8], qshift[8], kscale[8], kshift[8];
#pragma unroll
    for (int j = 0; j < 8; ++j) {
        int ch = mbase + wv * 16 + j + 8 * lhi;
        float s1 = qg[ch] * rsqrtf(qv[ch] + 1e-5f);
        qscale[j] = s1; qshift[j] = qb[ch] - qm[ch] * s1;
        float s2 = kg[ch] * rsqrtf(kv[ch] + 1e-5f);
        kscale[j] = s2; kshift[j] = kb_[ch] - km[ch] * s2;
    }

    float vq[2][8], vk[2][8];             // LIF state, carried across passes
#pragma unroll
    for (int s = 0; s < 2; ++s)
#pragma unroll
        for (int j = 0; j < 8; ++j) { vq[s][j] = 0.f; vk[s][j] = 0.f; }

    for (int tp = 0; tp < 2; ++tp) {      // two timesteps per pass
        const v8f z = {};
        v8f accq[2][2], acck[2][2];
        accq[0][0] = z; accq[0][1] = z; accq[1][0] = z; accq[1][1] = z;
        acck[0][0] = z; acck[0][1] = z; acck[1][0] = z; acck[1][1] = z;

        for (int kb = 0; kb < KBLK; ++kb) {
            __syncthreads();
            // ---- stage B tiles: 8-byte copies, no cvt, no branches ----
#pragma unroll
            for (int tt = 0; tt < 2; ++tt) {
                int t = tp * 2 + tt;
                const _Float16* src =
                    xp + (((size_t)t * BB + b) * NP + n) * CC + kb * KTILE + k0s;
#if defined(HAVE_ASYNC_LDS)
                __builtin_amdgcn_global_load_async_to_lds_b64(
                    (g_b64_t)src, (l_b64_t)&lx[tt][nloc][k0s], 0, 0);
#else
                *(v4h*)&lx[tt][nloc][k0s] = *(const v4h*)src;
#endif
            }
#if defined(HAVE_ASYNC_LDS)
            __builtin_amdgcn_s_wait_asynccnt(0);
#endif
            __syncthreads();

            // ---- A fragments straight from pre-swizzled global weights ----
            size_t wbase = ((((size_t)mt * KBLK + kb) * 8 + wv) * 32 + lane) * 16;
            v16h aq = cat8(*(const v8h*)(wq_sw + wbase),
                           *(const v8h*)(wq_sw + wbase + 8));
            v16h ak = cat8(*(const v8h*)(wk_sw + wbase),
                           *(const v8h*)(wk_sw + wbase + 8));
            if (kb + 1 < KBLK) {
                __builtin_prefetch(wq_sw + wbase + 8 * 32 * 16, 0, 1);
                __builtin_prefetch(wk_sw + wbase + 8 * 32 * 16, 0, 1);
            }

#pragma unroll
            for (int tt = 0; tt < 2; ++tt) {
#pragma unroll
                for (int s = 0; s < 2; ++s) {
                    const _Float16* bp0 = &lx[tt][s * 16 + lrow][lhi * 8];
                    v16h bx = cat8(*(const v8h*)bp0, *(const v8h*)(bp0 + 16));
                    accq[tt][s] = __builtin_amdgcn_wmma_f32_16x16x32_f16(
                        false, aq, false, bx, (short)0, accq[tt][s], false, false);
                    acck[tt][s] = __builtin_amdgcn_wmma_f32_16x16x32_f16(
                        false, ak, false, bx, (short)0, acck[tt][s], false, false);
                }
            }
        }

        // ---- BN + LIF (sequential in t) + vectorized spike stores ----
#pragma unroll
        for (int tt = 0; tt < 2; ++tt) {
            int t = tp * 2 + tt;
#pragma unroll
            for (int s = 0; s < 2; ++s) {
                int ns = nbase + s * 16 + lrow;
                bool ok = (ns < NN);
                v8h pq, pk;
#pragma unroll
                for (int j = 0; j < 8; ++j) {
                    float sq = lif_step(vq[s][j], accq[tt][s][j] * qscale[j] + qshift[j]);
                    float sk = lif_step(vk[s][j], acck[tt][s][j] * kscale[j] + kshift[j]);
                    pq[j] = (_Float16)(ok ? sq : 0.f);   // pad region stores zeros
                    pk[j] = (_Float16)(ok ? sk : 0.f);
                }
                size_t o = (((size_t)t * BB + b) * NP + ns) * CC
                         + mbase + wv * 16 + 8 * lhi;
                *(v8h*)&qs[o]  = pq;                      // one b128 store each
                *(v8h*)&ksp[o] = pk;
            }
        }
    }
}

// ---------------------------------------------------------------------------
// Kernel 2: attn = LIF_t( sum_dh q )  -> [T,B,HEADS,NP] f32.
// q is [t][b][n][c]: the 48-channel sum is 6 contiguous v8h loads.
// ---------------------------------------------------------------------------
__global__ __launch_bounds__(256)
void attn_kernel(const _Float16* __restrict__ qs, float* __restrict__ attn)
{
    int idx = blockIdx.x * 256 + threadIdx.x;   // h*NP + n, 8*224 = 1792
    if (idx >= HEADS * NP) return;
    int h = idx / NP, n = idx % NP;
    int b = blockIdx.y;
    float v = 0.f;
    for (int t = 0; t < TT; ++t) {
        const _Float16* base = qs + (((size_t)t * BB + b) * NP + n) * CC + h * DH;
        float sum = 0.f;
#pragma unroll
        for (int g = 0; g < 6; ++g) {
            v8h q8 = *(const v8h*)(base + g * 8);
#pragma unroll
            for (int j = 0; j < 8; ++j) sum += (float)q8[j];
        }
        attn[(((size_t)t * BB + b) * HEADS + h) * NP + n] = lif_step(v, sum);
    }
}

// ---------------------------------------------------------------------------
// Kernel 3: out = LIF( BN( Wp*(attn .* k) + bp ) ). y built during staging
// (exact {0,1} in f16; k read as contiguous v4h). Output fp32 [T,B,C,N].
// ---------------------------------------------------------------------------
__global__ __launch_bounds__(256)
void proj_gemm_lif(const _Float16* __restrict__ ksp,
                   const float* __restrict__ attn,
                   const _Float16* __restrict__ wp_sw,
                   const float* __restrict__ bp,
                   const float* __restrict__ pg, const float* __restrict__ pb,
                   const float* __restrict__ pm, const float* __restrict__ pv,
                   float* __restrict__ out)
{
    __shared__ _Float16 ly[2][NTILE][LXPAD];

    const int tid  = threadIdx.x;
    const int lane = tid & 31;
    const int wv   = tid >> 5;
    const int b    = blockIdx.y;
    const int mt   = blockIdx.x / NBLK;
    const int nt   = blockIdx.x % NBLK;
    const int mbase = mt * MTILE;
    const int nbase = nt * NTILE;
    const int lrow = lane & 15;
    const int lhi  = lane >> 4;
    const int nloc = tid & 31;
    const int k0s  = (tid >> 5) * 4;
    const int n    = nbase + nloc;

    float pscale[8], pshift[8];
#pragma unroll
    for (int j = 0; j < 8; ++j) {
        int ch = mbase + wv * 16 + j + 8 * lhi;
        float s1 = pg[ch] * rsqrtf(pv[ch] + 1e-5f);
        pscale[j] = s1;
        pshift[j] = (bp[ch] - pm[ch]) * s1 + pb[ch];   // bias folded into shift
    }

    float vm[2][8];
#pragma unroll
    for (int s = 0; s < 2; ++s)
#pragma unroll
        for (int j = 0; j < 8; ++j) vm[s][j] = 0.f;

    for (int tp = 0; tp < 2; ++tp) {
        const v8f z = {};
        v8f acc[2][2];
        acc[0][0] = z; acc[0][1] = z; acc[1][0] = z; acc[1][1] = z;

        for (int kb = 0; kb < KBLK; ++kb) {
            const int cbase = kb * KTILE + k0s;
            const int h = cbase / DH;        // constant over the 4-run (48%4==0)
            __syncthreads();
#pragma unroll
            for (int tt = 0; tt < 2; ++tt) {
                int t = tp * 2 + tt;
                float a = attn[(((size_t)t * BB + b) * HEADS + h) * NP + n];
                v4h kv4 = *(const v4h*)(ksp + (((size_t)t * BB + b) * NP + n) * CC + cbase);
                v4h hv;
#pragma unroll
                for (int j = 0; j < 4; ++j)
                    hv[j] = (_Float16)(a * (float)kv4[j]);
                *(v4h*)&ly[tt][nloc][k0s] = hv;
            }
            __syncthreads();

            size_t wbase = ((((size_t)mt * KBLK + kb) * 8 + wv) * 32 + lane) * 16;
            v16h aw = cat8(*(const v8h*)(wp_sw + wbase),
                           *(const v8h*)(wp_sw + wbase + 8));
            if (kb + 1 < KBLK)
                __builtin_prefetch(wp_sw + wbase + 8 * 32 * 16, 0, 1);

#pragma unroll
            for (int tt = 0; tt < 2; ++tt) {
#pragma unroll
                for (int s = 0; s < 2; ++s) {
                    const _Float16* bp0 = &ly[tt][s * 16 + lrow][lhi * 8];
                    v16h by = cat8(*(const v8h*)bp0, *(const v8h*)(bp0 + 16));
                    acc[tt][s] = __builtin_amdgcn_wmma_f32_16x16x32_f16(
                        false, aw, false, by, (short)0, acc[tt][s], false, false);
                }
            }
        }

#pragma unroll
        for (int tt = 0; tt < 2; ++tt) {
            int t = tp * 2 + tt;
#pragma unroll
            for (int s = 0; s < 2; ++s) {
                int ns = nbase + s * 16 + lrow;
#pragma unroll
                for (int j = 0; j < 8; ++j) {
                    int ch = mbase + wv * 16 + j + 8 * lhi;
                    float sp = lif_step(vm[s][j], acc[tt][s][j] * pscale[j] + pshift[j]);
                    if (ns < NN)
                        out[(((size_t)t * BB + b) * CC + ch) * NN + ns] = sp;
                }
            }
        }
    }
}

// ---------------------------------------------------------------------------
extern "C" void kernel_launch(void* const* d_in, const int* in_sizes, int n_in,
                              void* d_out, int out_size, void* d_ws, size_t ws_size,
                              hipStream_t stream) {
    const float* x  = (const float*)d_in[0];
    const float* Wq = (const float*)d_in[1];
    const float* qg = (const float*)d_in[2];
    const float* qb = (const float*)d_in[3];
    const float* qm = (const float*)d_in[4];
    const float* qv = (const float*)d_in[5];
    const float* Wk = (const float*)d_in[6];
    const float* kg = (const float*)d_in[7];
    const float* kb = (const float*)d_in[8];
    const float* km = (const float*)d_in[9];
    const float* kv = (const float*)d_in[10];
    const float* Wp = (const float*)d_in[11];
    const float* bp = (const float*)d_in[12];
    const float* pg = (const float*)d_in[13];
    const float* pb = (const float*)d_in[14];
    const float* pm = (const float*)d_in[15];
    const float* pv = (const float*)d_in[16];
    float* out = (float*)d_out;

    const size_t XPN = (size_t)TT * BB * NP * CC;   // padded, transposed
    _Float16* wq_sw = (_Float16*)d_ws;
    _Float16* wk_sw = wq_sw + WSWZ;
    _Float16* wp_sw = wk_sw + WSWZ;
    _Float16* xp    = wp_sw + WSWZ;
    _Float16* qs    = xp + XPN;
    _Float16* ksp   = qs + XPN;
    float*    attn  = (float*)(ksp + XPN);          // [T,B,HEADS,NP]

    dim3 swz_grid((unsigned)(WSWZ / 256));
    swizzle_w<<<swz_grid, 256, 0, stream>>>(Wq, wq_sw);
    swizzle_w<<<swz_grid, 256, 0, stream>>>(Wk, wk_sw);
    swizzle_w<<<swz_grid, 256, 0, stream>>>(Wp, wp_sw);

    dim3 xp_grid(TT * BB, NBLK * KBLK);             // (256, 84)
    xpack_kernel<<<xp_grid, 256, 0, stream>>>(x, xp);

    dim3 gemm_grid(MBLK * NBLK, BB);
    qk_gemm_lif<<<gemm_grid, 256, 0, stream>>>(xp, wq_sw, wk_sw,
                                               qg, qb, qm, qv,
                                               kg, kb, km, kv, qs, ksp);

    dim3 attn_grid((HEADS * NP + 255) / 256, BB);   // (7, 64)
    attn_kernel<<<attn_grid, 256, 0, stream>>>(qs, attn);

    proj_gemm_lif<<<gemm_grid, 256, 0, stream>>>(ksp, attn, wp_sw, bp,
                                                 pg, pb, pm, pv, out);
}
